// MultiGraphAttentionWithConcat_56418690400268
// MI455X (gfx1250) — compile-verified, matrix operand order checked by reading, not run on hardware
//
#include <hip/hip_runtime.h>

// ---------------------------------------------------------------------------
// MultiGraphAttentionWithConcat for MI455X (gfx1250, wave32, WMMA bf16)
// B=4 Lq=512 G=4 Lkv=1024 dim=dim_kv=inner=1024 heads=16 d=64
// ---------------------------------------------------------------------------

typedef __bf16 bf16_t;
typedef __attribute__((ext_vector_type(16))) __bf16 v16bf;
typedef __attribute__((ext_vector_type(8)))  __bf16 v8bf;
typedef __attribute__((ext_vector_type(8)))  float  v8f;

static __device__ __forceinline__ bf16_t f2bf(float f) {
  return (bf16_t)f;  // native f32->bf16 convert (RNE)
}

static __device__ __forceinline__ v8f wmma_bf16(v16bf a, v16bf b, v8f c) {
  // D = A(16x32) x B(32x16) + C, fp32 accumulate
  return __builtin_amdgcn_wmma_f32_16x16x32_bf16(false, a, false, b, (short)0, c,
                                                 false, false);
}

// Load one 16x32 bf16 WMMA operand tile from a row-major array.
// A-layout: lane L -> row row0 + (L&15); K chunks [k0+8*(L>>4), +7] and [+16..+23].
// (B operand uses the mirrored layout: "row" = output column index.)
static __device__ __forceinline__ v16bf load_tile(const bf16_t* base, long row0,
                                                  int ld, long k0, int lane) {
  const bf16_t* p = base + (row0 + (lane & 15)) * (long)ld + k0 + ((lane >> 4) << 3);
  union { v16bf v; v8bf h[2]; } u;
  u.h[0] = *(const v8bf*)(p);
  u.h[1] = *(const v8bf*)(p + 16);
  return u.v;
}

// ------------------------------ prep kernels -------------------------------

__global__ void k_cvt_bf16(const float* __restrict__ src, bf16_t* __restrict__ dst,
                           int n) {
  int i = blockIdx.x * blockDim.x + threadIdx.x;
  if (i < n) dst[i] = f2bf(src[i]);
}

__global__ void k_maskadd(const int* __restrict__ m, float* __restrict__ out, int n) {
  int i = blockIdx.x * blockDim.x + threadIdx.x;
  if (i < n) out[i] = m[i] ? 0.0f : -1e30f;
}

// WT[n][k] = bf16(W[k][n]), W is 1024x1024 fp32. Block (32,8), grid (32,32).
__global__ void k_transpose_cvt(const float* __restrict__ W, bf16_t* __restrict__ WT) {
  __shared__ float tile[32][33];
  const int bx = blockIdx.x * 32, by = blockIdx.y * 32;
  const int tx = threadIdx.x, ty = threadIdx.y;
#pragma unroll
  for (int i = 0; i < 32; i += 8)
    tile[ty + i][tx] = W[(long)(by + ty + i) * 1024 + bx + tx];
  __syncthreads();
#pragma unroll
  for (int i = 0; i < 32; i += 8)
    WT[(long)(bx + ty + i) * 1024 + by + tx] = f2bf(tile[tx][ty + i]);
}

// ------------------------------- GEMM --------------------------------------
// Per 16x16 tile: C[m][n] = sum_k A[m][k]*B[n][k]  (A,B row-major bf16, K=1024).
// Wave computes a 64x64 block: 16 accumulators, 16 loads : 16 WMMA per K-step.
// Transposed store => contiguous along m.
// MODE 0: D bf16, off = n*1024 + m                      (Q, K projections)
// MODE 1: D bf16, off = (m>>10)<<20 | n<<10 | (m&1023)  (V transposed per (b,g))
// MODE 2: D fp32, out[b][lq][g*1024+m], n = (b*4+g)*512+lq  (output proj)

template <int MODE>
static __device__ __forceinline__ void store_tile(void* Dp, long mt, long nt, v8f c,
                                                  float s, int lane) {
  const long n = nt + (lane & 15);
  const long m = mt + ((lane >> 4) << 3);
  if (MODE == 0 || MODE == 1) {
    v8bf h;
#pragma unroll
    for (int i = 0; i < 8; ++i) h[i] = f2bf(c[i] * s);
    long off = (MODE == 0) ? (n * 1024 + m)
                           : (((m >> 10) << 20) + (n << 10) + (m & 1023));
    *(v8bf*)((bf16_t*)Dp + off) = h;
  } else {
    const long b = n >> 11, g = (n >> 9) & 3, lq = n & 511;
    float* p = (float*)Dp + b * 2097152 + lq * 4096 + g * 1024 + m;
    float4 f0 = make_float4(c[0] * s, c[1] * s, c[2] * s, c[3] * s);
    float4 f1 = make_float4(c[4] * s, c[5] * s, c[6] * s, c[7] * s);
    *(float4*)p = f0;
    *(float4*)(p + 4) = f1;
  }
}

template <int MODE>
__global__ void k_gemm(const bf16_t* __restrict__ A, const bf16_t* __restrict__ B,
                       void* __restrict__ Dp, int M, int N, float scale) {
  const int lane = threadIdx.x & 31;
  const int wid = (blockIdx.x * blockDim.x + threadIdx.x) >> 5;
  const int tilesM = M >> 6, tilesN = N >> 6;
  if (wid >= tilesM * tilesN) return;
  const long m0 = (long)(wid % tilesM) * 64;
  const long n0 = (long)(wid / tilesM) * 64;

  v8f acc[4][4];
#pragma unroll
  for (int i = 0; i < 4; ++i)
#pragma unroll
    for (int j = 0; j < 4; ++j) acc[i][j] = (v8f){};

  for (int k = 0; k < 1024; k += 32) {
    v16bf a[4], b[4];
#pragma unroll
    for (int i = 0; i < 4; ++i) a[i] = load_tile(A, m0 + i * 16, 1024, k, lane);
#pragma unroll
    for (int j = 0; j < 4; ++j) b[j] = load_tile(B, n0 + j * 16, 1024, k, lane);
#pragma unroll
    for (int i = 0; i < 4; ++i)
#pragma unroll
      for (int j = 0; j < 4; ++j) acc[i][j] = wmma_bf16(a[i], b[j], acc[i][j]);
  }
#pragma unroll
  for (int i = 0; i < 4; ++i)
#pragma unroll
    for (int j = 0; j < 4; ++j)
      store_tile<MODE>(Dp, m0 + i * 16, n0 + j * 16, acc[i][j], scale, lane);
}

// ----------------------------- attention -----------------------------------
// One wave handles (b,g,h, 16 queries). Computes scores^T = K.q^T so softmax is
// in-lane; flash-style online softmax over Lkv=1024 in 32-row blocks; ctx^T
// accumulated with Vt as A operand and exp'd probabilities packed in-lane as B.

__global__ void k_attn(const bf16_t* __restrict__ Qb, const bf16_t* __restrict__ Kb,
                       const bf16_t* __restrict__ Vt, const float* __restrict__ Ma,
                       bf16_t* __restrict__ Ctx) {
  const int lane = threadIdx.x & 31;
  const int wid = (blockIdx.x * blockDim.x + threadIdx.x) >> 5;
  if (wid >= 8192) return;
  const int qt = wid & 31;          // 16-query tile, 0..31
  const int h  = (wid >> 5) & 15;   // head
  const int bg = wid >> 9;          // b*4+g, 0..15
  const int b  = bg >> 2;

  const long hcol  = (long)h * 64;
  const long qrow0 = (long)b * 512 + (long)qt * 16;
  const v16bf qb0 = load_tile(Qb, qrow0, 1024, hcol,      lane);
  const v16bf qb1 = load_tile(Qb, qrow0, 1024, hcol + 32, lane);

  const bf16_t* Kbg = Kb + (long)bg * 1024 * 1024;
  const bf16_t* Vbg = Vt + (long)bg * 1024 * 1024;
  const float*  Mbg = Ma + (long)bg * 1024;
  const int i0 = (lane >> 4) << 3;  // 0 or 8: which kv sub-rows this half owns

  v8f o0 = {}, o1 = {}, o2 = {}, o3 = {};
  float mrun = -1e30f, lrun = 0.0f;

  for (int k0 = 0; k0 < 1024; k0 += 32) {
    // scores^T: M = kv row, N = query. Two 16x16 tiles (kv k0.. / k0+16..)
    v8f st0 = {}, st1 = {};
    st0 = wmma_bf16(load_tile(Kbg, k0,      1024, hcol,      lane), qb0, st0);
    st0 = wmma_bf16(load_tile(Kbg, k0,      1024, hcol + 32, lane), qb1, st0);
    st1 = wmma_bf16(load_tile(Kbg, k0 + 16, 1024, hcol,      lane), qb0, st1);
    st1 = wmma_bf16(load_tile(Kbg, k0 + 16, 1024, hcol + 32, lane), qb1, st1);

    // additive mask per kv row (rows r+i0 of each tile live in this half)
    const float4 ma0 = *(const float4*)(Mbg + k0 + i0);
    const float4 ma1 = *(const float4*)(Mbg + k0 + i0 + 4);
    const float4 ma2 = *(const float4*)(Mbg + k0 + 16 + i0);
    const float4 ma3 = *(const float4*)(Mbg + k0 + 16 + i0 + 4);
    st0[0] += ma0.x; st0[1] += ma0.y; st0[2] += ma0.z; st0[3] += ma0.w;
    st0[4] += ma1.x; st0[5] += ma1.y; st0[6] += ma1.z; st0[7] += ma1.w;
    st1[0] += ma2.x; st1[1] += ma2.y; st1[2] += ma2.z; st1[3] += ma2.w;
    st1[4] += ma3.x; st1[5] += ma3.y; st1[6] += ma3.z; st1[7] += ma3.w;

    // online softmax per query column (in-lane over 8 VGPRs + xor-16 combine)
    float mc = st0[0];
#pragma unroll
    for (int i = 1; i < 8; ++i) mc = fmaxf(mc, st0[i]);
#pragma unroll
    for (int i = 0; i < 8; ++i) mc = fmaxf(mc, st1[i]);
    mc = fmaxf(mc, __shfl_xor(mc, 16, 32));
    const float mnew = fmaxf(mrun, mc);
    const float f = __expf(mrun - mnew);

    float psum = 0.0f;
    v16bf pb;  // B operand for ctx^T: lane = query col, kv packed along VGPRs
#pragma unroll
    for (int i = 0; i < 8; ++i) {
      const float p0 = __expf(st0[i] - mnew);
      const float p1 = __expf(st1[i] - mnew);
      psum += p0 + p1;
      pb[i]     = f2bf(p0);   // kv = k0 + i0 + i
      pb[i + 8] = f2bf(p1);   // kv = k0 + 16 + i0 + i
    }
    psum += __shfl_xor(psum, 16, 32);
    lrun = lrun * f + psum;
    mrun = mnew;

#pragma unroll
    for (int i = 0; i < 8; ++i) { o0[i] *= f; o1[i] *= f; o2[i] *= f; o3[i] *= f; }
    // ctx^T += Vt(16d x 32kv) x P^T(32kv x 16q)
    o0 = wmma_bf16(load_tile(Vbg, hcol +  0, 1024, k0, lane), pb, o0);
    o1 = wmma_bf16(load_tile(Vbg, hcol + 16, 1024, k0, lane), pb, o1);
    o2 = wmma_bf16(load_tile(Vbg, hcol + 32, 1024, k0, lane), pb, o2);
    o3 = wmma_bf16(load_tile(Vbg, hcol + 48, 1024, k0, lane), pb, o3);
  }

  const float rl = (lrun > 0.0f) ? (1.0f / lrun) : 0.0f;
  // transposed store of ctx^T tiles == row-major ctx[(b,g,lq)][inner], bf16
  bf16_t* crow = Ctx + ((long)bg * 512 + (long)qt * 16 + (lane & 15)) * 1024 + hcol + i0;
  v8f* oc[4] = {&o0, &o1, &o2, &o3};
#pragma unroll
  for (int t = 0; t < 4; ++t) {
    v8bf hh;
#pragma unroll
    for (int i = 0; i < 8; ++i) hh[i] = f2bf((*oc[t])[i] * rl);
    *(v8bf*)(crow + t * 16) = hh;
  }
}

// ------------------------------- launch ------------------------------------

extern "C" void kernel_launch(void* const* d_in, const int* in_sizes, int n_in,
                              void* d_out, int out_size, void* d_ws, size_t ws_size,
                              hipStream_t stream) {
  const float* H   = (const float*)d_in[0];  // [4,512,1024]
  const float* KV  = (const float*)d_in[1];  // [4,4,1024,1024]
  const int*   KVm = (const int*)  d_in[2];  // [4,4,1024]
  const float* Wq  = (const float*)d_in[3];
  const float* Wk  = (const float*)d_in[4];
  const float* Wv  = (const float*)d_in[5];
  const float* Wo  = (const float*)d_in[6];
  float* out = (float*)d_out;

  char* ws = (char*)d_ws;
  const size_t MiB = (size_t)1 << 20;
  bf16_t* Hb  = (bf16_t*)(ws + 0);         //  4 MiB: H bf16 [2048][1024]
  bf16_t* KVb = (bf16_t*)(ws + 4 * MiB);   // 32 MiB: KV bf16 [16384][1024]
  bf16_t* Ctx = KVb;                       // 16 MiB alias (KVb dead by then)
  bf16_t* WqT = (bf16_t*)(ws + 36 * MiB);  //  2 MiB each, W^T row-major bf16
  bf16_t* WkT = (bf16_t*)(ws + 38 * MiB);
  bf16_t* WvT = (bf16_t*)(ws + 40 * MiB);
  bf16_t* WoT = (bf16_t*)(ws + 42 * MiB);
  bf16_t* Qb  = (bf16_t*)(ws + 44 * MiB);  //  4 MiB: Q bf16 [2048][1024] (scaled)
  bf16_t* Kb  = (bf16_t*)(ws + 48 * MiB);  // 32 MiB: K bf16 [16384][1024]
  bf16_t* Vt  = (bf16_t*)(ws + 80 * MiB);  // 32 MiB: V^T bf16 [16][1024][1024]
  float*  Maf = (float*)(ws + 112 * MiB);  // 64 KiB: additive mask

  k_cvt_bf16<<<2097152 / 256, 256, 0, stream>>>(H, Hb, 2097152);
  k_cvt_bf16<<<16777216 / 256, 256, 0, stream>>>(KV, KVb, 16777216);
  k_maskadd<<<16384 / 256, 256, 0, stream>>>(KVm, Maf, 16384);

  dim3 tb(32, 8), tg(32, 32);
  k_transpose_cvt<<<tg, tb, 0, stream>>>(Wq, WqT);
  k_transpose_cvt<<<tg, tb, 0, stream>>>(Wk, WkT);
  k_transpose_cvt<<<tg, tb, 0, stream>>>(Wv, WvT);
  k_transpose_cvt<<<tg, tb, 0, stream>>>(Wo, WoT);

  // 64x64 blocks per wave: waves = (M/64)*(N/64), 4 waves (128 thr) per block
  // Q = 0.125 * H @ Wq   (row-major via (Wq^T, H) + transposed store)
  k_gemm<0><<<128, 128, 0, stream>>>(WqT, Hb, (void*)Qb, 1024, 2048, 0.125f);
  // K = KV @ Wk (row-major)
  k_gemm<0><<<1024, 128, 0, stream>>>(WkT, KVb, (void*)Kb, 1024, 16384, 1.0f);
  // V^T per (b,g): Vt[bg][inner][lkv]
  k_gemm<1><<<1024, 128, 0, stream>>>(KVb, WvT, (void*)Vt, 16384, 1024, 1.0f);
  // attention: 8192 waves = (b,g,h,qtile)
  k_attn<<<2048, 128, 0, stream>>>(Qb, Kb, Vt, Maf, Ctx);
  // out[b][lq][g*1024+n] = ctx @ Wo (fp32)
  k_gemm<2><<<512, 128, 0, stream>>>(WoT, Ctx, (void*)out, 1024, 8192, 1.0f);
}
